// SpatialBottleneck_46488726012553
// MI455X (gfx1250) — matrix operand order repaired and checked by reading, not compiled, not run on hardware
//
#include <hip/hip_runtime.h>
#include <hip/hip_bf16.h>

typedef __attribute__((ext_vector_type(16))) __bf16 v16bf;
typedef __attribute__((ext_vector_type(8)))  __bf16 v8bf;
typedef __attribute__((ext_vector_type(8)))  float  v8f;

// ---- WMMA fragment helpers (CDNA5 16x16x32 bf16 layouts, wave32) ----
// A (16xK): lane m = lane&15, kbA = lane<16?0:8, elements K = kbA+{0..7,16..23}
__device__ __forceinline__ v16bf frag_a(const __bf16* row, int k0, int kbA) {
    const __bf16* p = row + k0 + kbA;
    v8bf lo = *(const v8bf*)(p);
    v8bf hi = *(const v8bf*)(p + 16);
    v16bf r;
#pragma unroll
    for (int i = 0; i < 8; ++i) { r[i] = lo[i]; r[i + 8] = hi[i]; }
    return r;
}
// B (Kx16) from row-major B^T [n][k]: lane n = lane&15, K = kbB/2+{0..7,16..23}, kbB = lane<16?0:16
__device__ __forceinline__ v16bf frag_b(const __bf16* row, int k0, int kbB) {
    const __bf16* p = row + k0 + (kbB >> 1);
    v8bf lo = *(const v8bf*)(p);
    v8bf hi = *(const v8bf*)(p + 16);
    v16bf r;
#pragma unroll
    for (int i = 0; i < 8; ++i) { r[i] = lo[i]; r[i + 8] = hi[i]; }
    return r;
}

__device__ __forceinline__ v8f wmma_bf16(v16bf a, v16bf b, v8f c) {
    return __builtin_amdgcn_wmma_f32_16x16x32_bf16(false, a, false, b,
                                                   (short)0, c, false, false);
}

// Async global -> LDS copy, 16B per lane, tracked by ASYNCcnt (CDNA5).
__device__ __forceinline__ void async_g2l_b128(unsigned int lds_off, const void* gptr) {
    asm volatile("global_load_async_to_lds_b128 %0, %1, off"
                 :: "v"(lds_off), "v"((unsigned long long)(uintptr_t)gptr)
                 : "memory");
}
__device__ __forceinline__ void wait_async0() {
    asm volatile("s_wait_asynccnt 0" ::: "memory");
}

// -------- shared memory arena layout (bytes) --------
//  [0      ,  73728)  lw2   : bf16 [64][576]     (dead after conv2)
//  [73728  , 106496)  lw1   : bf16 [64][256]     (dead after conv1)
//  [106496 , 149504)  s_xt  : bf16 [336][64]     (dead after conv1)
//  [149504 , 182272)  lw3   : bf16 [256][64]
//  [182272 , 223744)  s_o1  : bf16 [324][64]
//  [223744 , 256512)  s_o2  : bf16 [256][64]
//  [256512 , 259584)  scale/bias floats (768)
//  conv3 residual double buffer (2 x 64KB fp32) overlays [0, 131072)
#define SMEM_BYTES 259584

__global__ __launch_bounds__(256, 1)
void bottleneck_fused(const float* __restrict__ xg,
                      const float* __restrict__ w1g, const float* __restrict__ w2g,
                      const float* __restrict__ w3g,
                      const float* __restrict__ s1g, const float* __restrict__ b1g,
                      const float* __restrict__ s2g, const float* __restrict__ b2g,
                      const float* __restrict__ s3g, const float* __restrict__ b3g,
                      float* __restrict__ outg)
{
    constexpr int H = 128, W = 128, Cin = 256, Cb = 64, Cout = 256;

    __shared__ __align__(16) unsigned char smem[SMEM_BYTES];
    __bf16* lw2  = (__bf16*)(smem);
    __bf16* lw1  = (__bf16*)(smem + 73728);
    __bf16* s_xt = (__bf16*)(smem + 106496);
    __bf16* lw3  = (__bf16*)(smem + 149504);
    __bf16* s_o1 = (__bf16*)(smem + 182272);
    __bf16* s_o2 = (__bf16*)(smem + 223744);
    float*  lsb  = (float*)(smem + 256512);
    float *ls1 = lsb, *lb1 = lsb + 64, *ls2 = lsb + 128, *lb2 = lsb + 192;
    float *ls3 = lsb + 256, *lb3 = lsb + 512;

    const int t      = threadIdx.x;
    const int lane   = t & 31;
    const int waveId = t >> 5;
    const int ml     = lane & 15;               // A row / B col / D col
    const int kbA    = (lane < 16) ? 0 : 8;
    const int kbB    = (lane < 16) ? 0 : 16;
    const int mrb    = (lane < 16) ? 0 : 8;     // D row base

    const int blk   = blockIdx.x;               // 16 imgs * 64 tiles
    const int n_img = blk >> 6;
    const int tloc  = blk & 63;
    const int h0    = (tloc >> 3) << 4;
    const int w0    = (tloc & 7) << 4;

    const v8f zero8 = {0.f, 0.f, 0.f, 0.f, 0.f, 0.f, 0.f, 0.f};

    // ---------------- stage 0: weights (fp32 -> bf16) + scale/bias into LDS ----
    for (int e = t; e < Cb * Cin; e += 256) lw1[e] = (__bf16)w1g[e];
    for (int e = t; e < Cb * 576; e += 256) {
        int c2 = e / 576, rem = e - c2 * 576;
        int tap = rem >> 6, c = rem & 63;
        lw2[e] = (__bf16)w2g[(c2 * 64 + c) * 9 + tap];
    }
    for (int e = t; e < Cout * Cb; e += 256) lw3[e] = (__bf16)w3g[e];
    if (t < 64) { ls1[t] = s1g[t]; lb1[t] = b1g[t]; ls2[t] = s2g[t]; lb2[t] = b2g[t]; }
    ls3[t] = s3g[t]; lb3[t] = b3g[t];
    for (int e = 324 * 64 + t; e < 336 * 64; e += 256) s_xt[e] = (__bf16)0.f; // pad tiles
    __syncthreads();

    // ---------------- conv1: 1x1 (256->64) over 18x18 halo, K streamed in 64-ch chunks
    v8f acc[11];
#pragma unroll
    for (int i = 0; i < 11; ++i) acc[i] = zero8;

    for (int c0 = 0; c0 < Cin; c0 += 64) {
        // stage x chunk, transposed to [pixel][channel] bf16 (zero-pad outside image)
        for (int e = t; e < 324 * 64; e += 256) {
            int c = e / 324;
            int hp = e - c * 324;
            int hr = hp / 18, hc = hp - hr * 18;
            int gh = h0 + hr - 1, gw = w0 + hc - 1;
            float v = 0.f;
            if ((unsigned)gh < (unsigned)H && (unsigned)gw < (unsigned)W)
                v = xg[(((size_t)n_img * Cin + (c0 + c)) * H + gh) * W + gw];
            s_xt[hp * 64 + c] = (__bf16)v;
        }
        __syncthreads();
#pragma unroll
        for (int kk = 0; kk < 64; kk += 32) {
#pragma unroll
            for (int i = 0; i < 11; ++i) {
                int idx = waveId + i * 8;           // 84 tiles = 4 Mtiles x 21 Ntiles
                if (idx < 84) {                     // wave-uniform guard (EXEC stays full)
                    int mt = idx / 21, nt = idx - mt * 21;
                    v16bf a = frag_a(&lw1[(mt * 16 + ml) * Cin + c0], kk, kbA);
                    v16bf b = frag_b(&s_xt[(nt * 16 + ml) * 64], kk, kbB);
                    acc[i] = wmma_bf16(a, b, acc[i]);
                }
            }
        }
        __syncthreads();
    }
    // epilogue conv1: scale/bias + relu -> bf16 o1[haloPix][c1]
#pragma unroll
    for (int i = 0; i < 11; ++i) {
        int idx = waveId + i * 8;
        if (idx < 84) {
            int mt = idx / 21, nt = idx - mt * 21;
            int pix = nt * 16 + ml;
            if (pix < 324) {
#pragma unroll
                for (int r = 0; r < 8; ++r) {
                    int c = mt * 16 + mrb + r;
                    float v = fmaxf(acc[i][r] * ls1[c] + lb1[c], 0.f);
                    s_o1[pix * 64 + c] = (__bf16)v;
                }
            }
        }
    }
    __syncthreads();

    // ---------------- conv2: 3x3 (64->64), implicit GEMM K = 9 taps x 64
    v8f acc2[8];
#pragma unroll
    for (int i = 0; i < 8; ++i) acc2[i] = zero8;
#pragma unroll
    for (int tap = 0; tap < 9; ++tap) {
        const int dy = tap / 3, dx = tap - dy * 3;
#pragma unroll
        for (int kk = 0; kk < 64; kk += 32) {
#pragma unroll
            for (int i = 0; i < 8; ++i) {           // 64 tiles = 4 Mtiles x 16 Ntiles
                int idx = waveId + i * 8;
                int mt = idx >> 4, nt = idx & 15;   // nt == local output row
                v16bf a = frag_a(&lw2[(mt * 16 + ml) * 576 + tap * 64], kk, kbA);
                v16bf b = frag_b(&s_o1[((nt + dy) * 18 + ml + dx) * 64], kk, kbB);
                acc2[i] = wmma_bf16(a, b, acc2[i]);
            }
        }
    }
#pragma unroll
    for (int i = 0; i < 8; ++i) {
        int idx = waveId + i * 8;
        int mt = idx >> 4, nt = idx & 15;
        int pix = nt * 16 + ml;
#pragma unroll
        for (int r = 0; r < 8; ++r) {
            int c = mt * 16 + mrb + r;
            float v = fmaxf(acc2[i][r] * ls2[c] + lb2[c], 0.f);
            s_o2[pix * 64 + c] = (__bf16)v;
        }
    }
    __syncthreads();   // also retires all reads of lw2/lw1/s_xt (overlaid below)

    // ---------------- conv3: 1x1 (64->256) in 4 channel-groups of 64,
    // residual double-buffered via async global->LDS (ASYNCcnt) overlapping WMMA.
    const size_t       img_base  = ((size_t)n_img * Cout) << 14;   // * H * W
    const unsigned int smem_base = (unsigned int)(uintptr_t)smem;  // LDS byte offset

    // prefetch residual group 0 -> buffer 0
#pragma unroll
    for (int j = 0; j < 16; ++j) {
        int e = t + j * 256;                    // 4096 x b128 = 64KB
        int c_local = e >> 6;
        int v4 = e & 63;
        int lr = v4 >> 2, lc4 = (v4 & 3) << 2;
        const float* g = xg + img_base + ((size_t)c_local << 14)
                       + (size_t)(h0 + lr) * W + (w0 + lc4);
        async_g2l_b128(smem_base + e * 16, g);
    }
    wait_async0();
    __syncthreads();

    for (int cg = 0; cg < 4; ++cg) {
        if (cg < 3) {  // prefetch next group into the other buffer while we compute
#pragma unroll
            for (int j = 0; j < 16; ++j) {
                int e = t + j * 256;
                int c_local = e >> 6;
                int v4 = e & 63;
                int lr = v4 >> 2, lc4 = (v4 & 3) << 2;
                const float* g = xg + img_base + ((size_t)((cg + 1) * 64 + c_local) << 14)
                               + (size_t)(h0 + lr) * W + (w0 + lc4);
                async_g2l_b128(smem_base + (((cg + 1) & 1) << 16) + e * 16, g);
            }
        }
        const float* resb = (const float*)(smem + ((cg & 1) << 16));  // [c_local][pix]
#pragma unroll
        for (int i = 0; i < 8; ++i) {
            int idx = waveId + i * 8;           // 64 tiles = 4 local Mtiles x 16 Ntiles
            int mtl = idx >> 4, nt = idx & 15;
            int mt = cg * 4 + mtl;
            v8f c3 = zero8;
#pragma unroll
            for (int kk = 0; kk < 64; kk += 32) {
                v16bf a = frag_a(&lw3[(mt * 16 + ml) * 64], kk, kbA);
                v16bf b = frag_b(&s_o2[(nt * 16 + ml) * 64], kk, kbB);
                c3 = wmma_bf16(a, b, c3);
            }
            const int gh = h0 + nt, gw = w0 + ml;
#pragma unroll
            for (int r = 0; r < 8; ++r) {
                int c = mt * 16 + mrb + r;
                int c_local = (mtl << 4) + mrb + r;
                float resv = resb[c_local * 256 + nt * 16 + ml];
                float v = c3[r] * ls3[c] + lb3[c] + resv;
                size_t addr = img_base + ((size_t)c << 14) + (size_t)gh * W + gw;
                outg[addr] = fmaxf(v, 0.f);
            }
        }
        if (cg < 3) {
            wait_async0();       // next group's residual landed
            __syncthreads();     // and everyone is done reading the buffer we overwrite next
        }
    }
}

extern "C" void kernel_launch(void* const* d_in, const int* in_sizes, int n_in,
                              void* d_out, int out_size, void* d_ws, size_t ws_size,
                              hipStream_t stream) {
    const float* x  = (const float*)d_in[0];
    const float* w1 = (const float*)d_in[1];
    const float* w2 = (const float*)d_in[2];
    const float* w3 = (const float*)d_in[3];
    const float* s1 = (const float*)d_in[4];
    const float* b1 = (const float*)d_in[5];
    const float* s2 = (const float*)d_in[6];
    const float* b2 = (const float*)d_in[7];
    const float* s3 = (const float*)d_in[8];
    const float* b3 = (const float*)d_in[9];
    float* out = (float*)d_out;

    dim3 grid(16 * 64);   // N images x (128/16)^2 spatial tiles
    dim3 block(256);      // 8 wave32s
    bottleneck_fused<<<grid, block, 0, stream>>>(x, w1, w2, w3,
                                                 s1, b1, s2, b2, s3, b3, out);
}
